// FSQRVQVae_5093831213413
// MI455X (gfx1250) — matrix operand order repaired and checked by reading, not compiled, not run on hardware
//
#include <hip/hip_runtime.h>
#include <stdint.h>

// ---------------------------------------------------------------------------
// CDNA5 (gfx1250) FSQ-RVQ-VAE forward.
// Conv1d layers run as implicit GEMMs on v_wmma_f32_16x16x32_bf16 (wave32).
// Wave tile: 32 C_out x 32 T (4 accumulators, A/B fragment reuse).
// Block tile: 256 C_out x 32 T, double-buffered LDS staging (1 barrier/chunk).
// ---------------------------------------------------------------------------

typedef __attribute__((ext_vector_type(16))) __bf16 v16bf;
typedef __attribute__((ext_vector_type(8)))  float  v8f;

__device__ __forceinline__ uint16_t f2bf(float f) {
  uint32_t u = __float_as_uint(f);
  uint32_t r = (u + 0x7FFFu + ((u >> 16) & 1u)) >> 16;  // round-to-nearest-even
  return (uint16_t)r;
}
__device__ __forceinline__ float bf2f(uint16_t h) {
  return __uint_as_float(((uint32_t)h) << 16);
}

union ABFrag {
  v16bf    v;
  uint4    q[2];
  uint16_t u[16];
};

// ---------------------------------------------------------------------------
// Input cast: features f32 [b][t][133] -> bf16 [b][160][2048] (zero padded C)
// ---------------------------------------------------------------------------
__global__ void cast_in(const float* __restrict__ F, uint16_t* __restrict__ out) {
  size_t i = (size_t)blockIdx.x * 256 + threadIdx.x;
  const size_t total = (size_t)16 * 160 * 2048;
  if (i >= total) return;
  int t = (int)(i % 2048);
  int c = (int)((i / 2048) % 160);
  int b = (int)(i / (2048 * 160));
  float v = (c < 133) ? F[((size_t)b * 2048 + t) * 133 + c] : 0.0f;
  out[i] = f2bf(v);
}

// ---------------------------------------------------------------------------
// Weight pack: W f32 [O][I][K] -> bf16 Wp[k][Op][Ip], zero padded
// ---------------------------------------------------------------------------
__global__ void pack_w(const float* __restrict__ w, uint16_t* __restrict__ wp,
                       int O, int I, int K, int Op, int Ip) {
  int i = blockIdx.x * 256 + threadIdx.x;
  int total = K * Op * Ip;
  if (i >= total) return;
  int ci = i % Ip;
  int co = (i / Ip) % Op;
  int k  = i / (Ip * Op);
  float v = (co < O && ci < I) ? w[((size_t)co * I + ci) * K + k] : 0.0f;
  wp[i] = f2bf(v);
}

// ---------------------------------------------------------------------------
// Generic conv1d as implicit GEMM on WMMA bf16 -> f32 accumulate.
// 256 threads = 8 waves; wave computes 32co x 32t via 4 WMMAs per K-chunk.
// Supports stride, dilation, pad, input-ReLU, output-ReLU, residual add,
// repeat-by-2 upsample on the input, bf16 and/or f32 outputs.
// ---------------------------------------------------------------------------
__global__ __launch_bounds__(256) void conv1d_wmma(
    const uint16_t* __restrict__ X, const uint16_t* __restrict__ Wp,
    const float* __restrict__ bias,
    uint16_t* __restrict__ Ybf, float* __restrict__ Yf32,
    const uint16_t* __restrict__ resid,
    int Cin_pad, int Tin, int Cout, int Cout_pad, int Tout,
    int ks, int stride, int dil, int pad,
    int relu_in, int relu_out, int repeat2, int final_layout) {
  // double-buffered [t][ci] tile: 2 x 32t x 32ci bf16 = 4 KB
  __shared__ __align__(16) uint16_t lds[2][32][32];

  const int wave    = threadIdx.x >> 5;
  const int lane    = threadIdx.x & 31;
  const int t_base  = blockIdx.x * 32;
  const int co_base = blockIdx.y * 256 + wave * 32;
  const int b       = blockIdx.z;
  const int m       = lane & 15;
  const int half    = lane >> 4;

  v8f acc00, acc01, acc10, acc11;
#pragma unroll
  for (int i = 0; i < 8; ++i) {
    acc00[i] = 0.0f; acc01[i] = 0.0f; acc10[i] = 0.0f; acc11[i] = 0.0f;
  }

  const int    Tlim  = repeat2 ? (Tin << 1) : Tin;
  const size_t xbase = (size_t)b * Cin_pad * Tin;
  const int    coA0  = (co_base + m      < Cout_pad) ? (co_base + m)      : (Cout_pad - 1);
  const int    coA1  = (co_base + 16 + m < Cout_pad) ? (co_base + 16 + m) : (Cout_pad - 1);

  // staging coordinates (branchless, clause-friendly): thread owns one t
  // column and 4 ci rows per chunk.
  const int tn  = threadIdx.x & 31;   // 0..31 : t within tile
  const int ci0 = threadIdx.x >> 5;   // 0..7  : first ci row (step 8)

  int p = 0;
  for (int k = 0; k < ks; ++k) {
    const int toff = k * dil - pad;
    // per-tap t mapping for this thread's staging column
    int t = (t_base + tn) * stride + toff;
    const bool inb = (t >= 0) && (t < Tlim);
    int ts = repeat2 ? (t >> 1) : t;
    ts = inb ? ts : 0;                       // clamp: always-legal address
    const uint16_t* xp = X + xbase + ts;

    for (int cc = 0; cc < Cin_pad; cc += 32) {
      // -------- stage 32ci x 32t tile (transposed) into lds[p] --------------
#pragma unroll
      for (int j = 0; j < 4; ++j) {
        const int cl = ci0 + 8 * j;
        uint16_t v = xp[(size_t)(cc + cl) * Tin];
        v = inb ? v : (uint16_t)0;
        if (relu_in) v = (v & 0x8000u) ? (uint16_t)0 : v;
        lds[p][tn][cl] = v;
      }
      // prefetch next chunk's cachelines (global_prefetch_b8)
      if (cc + 32 < Cin_pad && wave == 0) {
        __builtin_prefetch(&xp[(size_t)(cc + 32 + lane) * Tin], 0, 1);
      }
      __syncthreads();

      // -------- A fragments: packed weights, ISA 16-bit A 16x32 layout ------
      ABFrag a0, a1, b0, b1;
      const uint4* ar0 =
          (const uint4*)(Wp + (((size_t)k * Cout_pad + coA0) * Cin_pad + cc));
      const uint4* ar1 =
          (const uint4*)(Wp + (((size_t)k * Cout_pad + coA1) * Cin_pad + cc));
      a0.q[0] = ar0[half];  a0.q[1] = ar0[2 + half];
      a1.q[0] = ar1[half];  a1.q[1] = ar1[2 + half];

      // -------- B fragments from LDS: lane = column, ISA 16-bit B 32x16 layout ----
      const uint4* br0 = (const uint4*)(&lds[p][m][0]);        // t = t_base+m
      const uint4* br1 = (const uint4*)(&lds[p][m + 16][0]);   // t = +16
      b0.q[0] = br0[half];  b0.q[1] = br0[2 + half];
      b1.q[0] = br1[half];  b1.q[1] = br1[2 + half];

      acc00 = __builtin_amdgcn_wmma_f32_16x16x32_bf16(false, a0.v, false, b0.v,
                                                      (short)0, acc00, false, false);
      acc01 = __builtin_amdgcn_wmma_f32_16x16x32_bf16(false, a0.v, false, b1.v,
                                                      (short)0, acc01, false, false);
      acc10 = __builtin_amdgcn_wmma_f32_16x16x32_bf16(false, a1.v, false, b0.v,
                                                      (short)0, acc10, false, false);
      acc11 = __builtin_amdgcn_wmma_f32_16x16x32_bf16(false, a1.v, false, b1.v,
                                                      (short)0, acc11, false, false);
      p ^= 1;  // double buffer: next stage writes the other tile
    }
  }

  // -------- epilogue: bias (+residual | ReLU), bf16/f32 stores ---------------
  const int n = lane & 15;
  const int t0 = t_base + n, t1 = t_base + 16 + n;
#pragma unroll
  for (int r = 0; r < 8; ++r) {
    const int cr0 = co_base + r + half * 8;       // C/D layout: reg r -> M=r(+8)
    const int cr1 = cr0 + 16;
    const float av[4] = {acc00[r], acc01[r], acc10[r], acc11[r]};
    const int   cos[4] = {cr0, cr0, cr1, cr1};
    const int   tss[4] = {t0, t1, t0, t1};
#pragma unroll
    for (int s = 0; s < 4; ++s) {
      const int co = cos[s], t = tss[s];
      if (co >= Cout_pad) continue;
      float val = 0.0f;
      if (co < Cout) {
        val = av[s] + bias[co];
        if (resid) val += bf2f(resid[((size_t)b * Cout_pad + co) * Tout + t]);
        if (relu_out && val < 0.0f) val = 0.0f;
      }
      if (Ybf) Ybf[((size_t)b * Cout_pad + co) * Tout + t] = f2bf(val);
      if (Yf32) {
        if (final_layout) {
          if (co < Cout) Yf32[((size_t)b * Tout + t) * Cout + co] = val;  // [b][t][c]
        } else {
          Yf32[((size_t)b * Cout_pad + co) * Tout + t] = val;
        }
      }
    }
  }
}

// ---------------------------------------------------------------------------
// One residual-FSQ iteration. LEVELS = [5,5,5,5] -> shift=0, half_l=1.998,
// half_width=2, basis {1,5,25,125}. One thread per (b,t).
// ---------------------------------------------------------------------------
__global__ void fsq_step(float* __restrict__ r, float* __restrict__ qsum,
                         int* __restrict__ counts) {
  int i = blockIdx.x * blockDim.x + threadIdx.x;
  if (i >= 16 * 256) return;
  int t = i & 255;
  int b = i >> 8;
  const int basis[4] = {1, 5, 25, 125};
  int idx = 0;
#pragma unroll
  for (int c = 0; c < 4; ++c) {
    size_t o = ((size_t)b * 16 + c) * 256 + t;
    float z = r[o];
    float bounded = tanhf(z) * 1.998f;
    float q = rintf(bounded);  // round half-to-even, matches jnp.round
    float qn = q * 0.5f;
    qsum[o] += qn;
    r[o] = z - qn;
    idx += (int)(q + 2.0f) * basis[c];
  }
  atomicAdd(&counts[idx], 1);
}

// Entropy -> perplexity, accumulated across the 6 quantizer stages.
__global__ void pp_kernel(const int* __restrict__ counts, float inv_n,
                          float* __restrict__ ppacc) {
  __shared__ float red[256];
  float s = 0.0f;
  for (int i = threadIdx.x; i < 625; i += 256) {
    float p = (float)counts[i] * inv_n;
    if (p > 0.0f) s += p * logf(p);
  }
  red[threadIdx.x] = s;
  __syncthreads();
  for (int k = 128; k > 0; k >>= 1) {
    if (threadIdx.x < k) red[threadIdx.x] += red[threadIdx.x + k];
    __syncthreads();
  }
  if (threadIdx.x == 0) atomicAdd(ppacc, expf(-red[0]));
}

// qsum f32 [b][16][256] -> decoder input bf16 [b][32][256] (zero padded C)
__global__ void qsum_to_decin(const float* __restrict__ qsum,
                              uint16_t* __restrict__ out) {
  size_t i = (size_t)blockIdx.x * 256 + threadIdx.x;
  const size_t total = (size_t)16 * 32 * 256;
  if (i >= total) return;
  int t = (int)(i % 256);
  int c = (int)((i / 256) % 32);
  int b = (int)(i / (256 * 32));
  float v = (c < 4) ? qsum[((size_t)b * 16 + c) * 256 + t] : 0.0f;
  out[i] = f2bf(v);
}

__global__ void finalize(float* __restrict__ out, const float* __restrict__ ppacc) {
  out[4358144] = 0.0f;                 // loss
  out[4358145] = ppacc[0] / 6.0f;      // mean perplexity over NUM_Q
}

// ---------------------------------------------------------------------------
// Host orchestration. Param order assumes jax tree_leaves flattening:
// dicts sorted alphabetically, lists in order; top level: features, params.
// params -> 'dec' (48 leaves) then 'enc' (46 leaves).
// ---------------------------------------------------------------------------
extern "C" void kernel_launch(void* const* d_in, const int* in_sizes, int n_in,
                              void* d_out, int out_size, void* d_ws, size_t ws_size,
                              hipStream_t stream) {
  (void)in_sizes; (void)out_size; (void)ws_size;
  if (n_in < 95) return;

  const int Bsz = 16, W = 512, NF = 133;
  const float* F = (const float*)d_in[0];
  auto P = [&](int i) { return (const float*)d_in[i]; };
  // dec leaves:  1 in.b  2 in.w  3 mid.b  4 mid.w  5 out.b  6 out.w
  //              up[u] @ 7+u*14: [b1,b2,w1,w2]x3, ub(+12), uw(+13)
  // enc leaves:  down[d] @ 49+d*14: db(+0), dw(+1), [b1,b2,w1,w2]x3
  //              91 in.b  92 in.w  93 out.b  94 out.w

  uint8_t* ws = (uint8_t*)d_ws;
  size_t off = 0;
  auto alloc = [&](size_t bytes) -> void* {
    void* p = ws + off;
    off = (off + bytes + 255) & ~(size_t)255;
    return p;
  };
  uint16_t* bufA  = (uint16_t*)alloc((size_t)16 * 512 * 2048 * 2);
  uint16_t* bufB  = (uint16_t*)alloc((size_t)16 * 512 * 2048 * 2);
  uint16_t* bufC  = (uint16_t*)alloc((size_t)16 * 512 * 2048 * 2);
  uint16_t* wpack = (uint16_t*)alloc((size_t)6 * 512 * 512 * 2);
  float*    zbuf  = (float*)alloc((size_t)16 * 16 * 256 * 4);
  float*    rbuf  = (float*)alloc((size_t)16 * 16 * 256 * 4);
  float*    qsum  = (float*)alloc((size_t)16 * 16 * 256 * 4);
  int*      counts = (int*)alloc((size_t)6 * 625 * 4);
  float*    ppacc  = (float*)alloc(256);
  uint16_t* decin  = (uint16_t*)alloc((size_t)16 * 32 * 256 * 2);

  auto conv = [&](const float* w, const float* bias, const uint16_t* X,
                  uint16_t* Ybf, float* Yf32, const uint16_t* resid,
                  int Cin, int Cin_pad, int Tin, int Cout, int Cout_pad, int Tout,
                  int ks, int stride, int dil, int pad,
                  int relu_in, int relu_out, int repeat2, int final_layout) {
    int total = ks * Cout_pad * Cin_pad;
    pack_w<<<(total + 255) / 256, 256, 0, stream>>>(w, wpack, Cout, Cin, ks,
                                                    Cout_pad, Cin_pad);
    dim3 g(Tout / 32, (Cout_pad + 255) / 256, Bsz);
    conv1d_wmma<<<g, 256, 0, stream>>>(X, wpack, bias, Ybf, Yf32, resid,
                                       Cin_pad, Tin, Cout, Cout_pad, Tout,
                                       ks, stride, dil, pad,
                                       relu_in, relu_out, repeat2, final_layout);
  };

  // ---------------- encoder ----------------
  {
    size_t total = (size_t)Bsz * 160 * 2048;
    cast_in<<<(unsigned)((total + 255) / 256), 256, 0, stream>>>(F, bufA);
  }
  conv(P(92), P(91), bufA, bufB, nullptr, nullptr, NF, 160, 2048, W, 512, 2048,
       3, 1, 1, 1, 0, 1, 0, 0);  // enc.in + relu
  uint16_t* x = bufB; uint16_t* s1 = bufA; uint16_t* s2 = bufC;
  int T = 2048;
  for (int d = 0; d < 3; ++d) {
    int base = 49 + d * 14;
    conv(P(base + 1), P(base + 0), x, s1, nullptr, nullptr, W, 512, T, W, 512,
         T / 2, 6, 2, 1, 1, 0, 0, 0, 0);  // strided down conv
    { uint16_t* tmp = x; x = s1; s1 = tmp; }
    T /= 2;
    for (int j = 0; j < 3; ++j) {
      int dil = 1; for (int q = 0; q < j; ++q) dil *= 3;  // 1,3,9
      const float* b1 = P(base + 2 + j * 4 + 0);
      const float* b2 = P(base + 2 + j * 4 + 1);
      const float* w1 = P(base + 2 + j * 4 + 2);
      const float* w2 = P(base + 2 + j * 4 + 3);
      conv(w1, b1, x, s2, nullptr, nullptr, W, 512, T, W, 512, T,
           3, 1, dil, dil, 1, 0, 0, 0);                    // h = conv(relu(x))
      conv(w2, b2, s2, x, nullptr, x, W, 512, T, W, 512, T,
           1, 1, 1, 0, 1, 0, 0, 0);                        // x += conv(relu(h))
    }
  }
  conv(P(94), P(93), x, nullptr, zbuf, nullptr, W, 512, T, 4, 16, T,
       3, 1, 1, 1, 0, 0, 0, 0);  // enc.out -> z (f32)

  // ---------------- residual FSQ ----------------
  const size_t zbytes = (size_t)Bsz * 16 * 256 * 4;
  hipMemcpyAsync(rbuf, zbuf, zbytes, hipMemcpyDeviceToDevice, stream);
  hipMemsetAsync(qsum, 0, zbytes, stream);
  hipMemsetAsync(counts, 0, (size_t)6 * 625 * 4, stream);
  hipMemsetAsync(ppacc, 0, 4, stream);
  for (int it = 0; it < 6; ++it) {
    fsq_step<<<(4096 + 255) / 256, 256, 0, stream>>>(rbuf, qsum,
                                                     counts + it * 625);
    pp_kernel<<<1, 256, 0, stream>>>(counts + it * 625, 1.0f / 4096.0f, ppacc);
  }
  {
    size_t total = (size_t)Bsz * 32 * 256;
    qsum_to_decin<<<(unsigned)((total + 255) / 256), 256, 0, stream>>>(qsum, decin);
  }

  // ---------------- decoder ----------------
  conv(P(2), P(1), decin, bufA, nullptr, nullptr, 4, 32, 256, W, 512, 256,
       3, 1, 1, 1, 0, 1, 0, 0);  // dec.in + relu
  x = bufA; s1 = bufB; s2 = bufC; T = 256;
  for (int u = 0; u < 3; ++u) {
    int base = 7 + u * 14;
    for (int j = 0; j < 3; ++j) {
      int dil = 1; for (int q = 0; q < 2 - j; ++q) dil *= 3;  // 9,3,1 (reversed)
      const float* b1 = P(base + j * 4 + 0);
      const float* b2 = P(base + j * 4 + 1);
      const float* w1 = P(base + j * 4 + 2);
      const float* w2 = P(base + j * 4 + 3);
      conv(w1, b1, x, s2, nullptr, nullptr, W, 512, T, W, 512, T,
           3, 1, dil, dil, 1, 0, 0, 0);
      conv(w2, b2, s2, x, nullptr, x, W, 512, T, W, 512, T,
           1, 1, 1, 0, 1, 0, 0, 0);
    }
    conv(P(base + 13), P(base + 12), x, s1, nullptr, nullptr, W, 512, T,
         W, 512, 2 * T, 3, 1, 1, 1, 0, 0, /*repeat2=*/1, 0);  // upsample+conv
    { uint16_t* tmp = x; x = s1; s1 = tmp; }
    T *= 2;
  }
  conv(P(4), P(3), x, s1, nullptr, nullptr, W, 512, T, W, 512, T,
       3, 1, 1, 1, 0, 1, 0, 0);  // mid + relu
  { uint16_t* tmp = x; x = s1; s1 = tmp; }
  conv(P(6), P(5), x, nullptr, (float*)d_out, nullptr, W, 512, T, 133, 144, T,
       3, 1, 1, 1, 0, 0, 0, /*final_layout=*/1);  // out -> [b][t][c] f32

  finalize<<<1, 1, 0, stream>>>((float*)d_out, ppacc);
}
